// Attn_36885179138293
// MI455X (gfx1250) — matrix operand order repaired and checked by reading, not compiled, not run on hardware
//
#include <hip/hip_runtime.h>

#define Bc 2
#define Hc 16
#define Sc 2048
#define Dc 128

#define KSTR 136   // K LDS row stride (bf16 elems), 272B: conflict-free b128 reads
#define VSTR 40    // V LDS (transposed) row stride, 80B: conflict-free
#define PSTR 40    // P staging row stride
#define PSZ  (16 * PSTR)
#define VROWS 144  // 128 d-rows + 16-row "ones" tile for row-sum accumulation

typedef __attribute__((ext_vector_type(16))) __bf16 v16bf;
typedef __attribute__((ext_vector_type(8)))  float  v8f;

union BFrag {            // 32 bytes: one WMMA bf16 A/B fragment per lane
    unsigned short s[16];
    uint4          u[2];
    v16bf          v;
};
union U8 {               // 16 bytes: 8 packed bf16
    unsigned short s[8];
    uint4          u;
};

// native bf16 conversion (hardware cvt path, RNE)
static __device__ __forceinline__ unsigned short f2bf(float x) {
    __bf16 b = (__bf16)x;
    return __builtin_bit_cast(unsigned short, b);
}

static __device__ __forceinline__ float4 ldg4(const float* p) {
    return *(const float4*)p;
}

__global__ __launch_bounds__(256) void alibi_attn_kernel(
    const float* __restrict__ Q, const float* __restrict__ K,
    const float* __restrict__ V, float* __restrict__ O) {

    __shared__ __align__(16) unsigned short sK[32 * KSTR];      // K block, row-major bf16
    __shared__ __align__(16) unsigned short sV[VROWS * VSTR];   // V block (transposed) + ones tile
    __shared__ __align__(16) unsigned short sP[8 * PSZ];        // per-wave P staging

    const int tid  = threadIdx.x;
    const int w    = tid >> 5;          // wave id 0..7
    const int lane = tid & 31;
    const int lm   = lane & 15;
    const int kh   = lane >> 4;

    const int qbase = blockIdx.x * 128;
    const int bh    = blockIdx.y;       // b*H + h
    const int h     = bh % Hc;
    const size_t base = (size_t)bh * Sc * Dc;

    const int   qrow0  = qbase + w * 16;
    const float slope  = exp2f(-(float)(h + 1) * (8.0f / (float)Hc));
    const float rscale = 0.08838834764831845f;   // 1/sqrt(128)

    // ---- one-time init of the "ones" tile (rows 128..143 of sV) ----
    for (int idx = tid; idx < 16 * 32; idx += 256) {
        const int d = 128 + (idx >> 5);
        const int c = idx & 31;
        sV[d * VSTR + c] = (d == 128) ? (unsigned short)0x3F80 : (unsigned short)0;
    }

    // ---- load Q fragments (A-layout, 16x32 bf16 per chunk, K=128 -> 4 chunks) ----
    const int qi = qrow0 + lm;
    const float* qp = Q + base + (size_t)qi * Dc;
    BFrag qf[4];
#pragma unroll
    for (int c = 0; c < 4; ++c) {
        const int k0 = c * 32 + kh * 8;
        float4 a0 = ldg4(qp + k0), a1 = ldg4(qp + k0 + 4);
        float4 b0 = ldg4(qp + k0 + 16), b1 = ldg4(qp + k0 + 20);
        qf[c].s[0] = f2bf(a0.x); qf[c].s[1] = f2bf(a0.y);
        qf[c].s[2] = f2bf(a0.z); qf[c].s[3] = f2bf(a0.w);
        qf[c].s[4] = f2bf(a1.x); qf[c].s[5] = f2bf(a1.y);
        qf[c].s[6] = f2bf(a1.z); qf[c].s[7] = f2bf(a1.w);
        qf[c].s[8]  = f2bf(b0.x); qf[c].s[9]  = f2bf(b0.y);
        qf[c].s[10] = f2bf(b0.z); qf[c].s[11] = f2bf(b0.w);
        qf[c].s[12] = f2bf(b1.x); qf[c].s[13] = f2bf(b1.y);
        qf[c].s[14] = f2bf(b1.z); qf[c].s[15] = f2bf(b1.w);
    }

    const v8f vzero = {0.f, 0.f, 0.f, 0.f, 0.f, 0.f, 0.f, 0.f};
    v8f acc[9];                    // 8 output d-tiles + rowsum tile (P @ ones)
#pragma unroll
    for (int j = 0; j < 9; ++j) acc[j] = vzero;
    float rmax[8];
#pragma unroll
    for (int r = 0; r < 8; ++r) rmax[r] = -1e30f;

    // ---- per-row ALiBi bias bases and causal thresholds (affine in n0) ----
    float bias0[8], bias1[8];
    int   thr0[8], thr1[8];
#pragma unroll
    for (int r = 0; r < 8; ++r) {
        const int i_r = qrow0 + r + kh * 8;
        bias0[r] = slope * (float)(lm - i_r);        // col = lm      (tile 0)
        bias1[r] = slope * (float)(16 + lm - i_r);   // col = 16 + lm (tile 1)
        thr0[r]  = i_r - lm;
        thr1[r]  = i_r - lm - 16;
    }
    const float bstep = slope * 32.0f;

    const int wmax  = qrow0 + 15;       // last valid KV row for this wave (causal)
    const int kvEnd = qbase + 128;

    // staging indices (cooperative, 256 threads over 32x128 block)
    const int sr   = tid >> 3;          // row within KV block 0..31
    const int dseg = (tid & 7) << 4;    // 16-wide d segment

    for (int n0 = 0; n0 < kvEnd; n0 += 32) {
        __syncthreads();   // previous iteration's LDS reads (and init) complete

        {   // ---- stage K (row-major bf16) and V (transposed bf16) ----
            const float* kp = K + base + (size_t)(n0 + sr) * Dc + dseg;
            const float* vp = V + base + (size_t)(n0 + sr) * Dc + dseg;
            float4 k0 = ldg4(kp), k1 = ldg4(kp + 4), k2 = ldg4(kp + 8), k3 = ldg4(kp + 12);
            U8 pa, pb;
            pa.s[0] = f2bf(k0.x); pa.s[1] = f2bf(k0.y); pa.s[2] = f2bf(k0.z); pa.s[3] = f2bf(k0.w);
            pa.s[4] = f2bf(k1.x); pa.s[5] = f2bf(k1.y); pa.s[6] = f2bf(k1.z); pa.s[7] = f2bf(k1.w);
            pb.s[0] = f2bf(k2.x); pb.s[1] = f2bf(k2.y); pb.s[2] = f2bf(k2.z); pb.s[3] = f2bf(k2.w);
            pb.s[4] = f2bf(k3.x); pb.s[5] = f2bf(k3.y); pb.s[6] = f2bf(k3.z); pb.s[7] = f2bf(k3.w);
            *(uint4*)&sK[sr * KSTR + dseg]     = pa.u;
            *(uint4*)&sK[sr * KSTR + dseg + 8] = pb.u;

            float4 v0 = ldg4(vp), v1 = ldg4(vp + 4), v2 = ldg4(vp + 8), v3 = ldg4(vp + 12);
            float vv[16] = {v0.x, v0.y, v0.z, v0.w, v1.x, v1.y, v1.z, v1.w,
                            v2.x, v2.y, v2.z, v2.w, v3.x, v3.y, v3.z, v3.w};
#pragma unroll
            for (int jj = 0; jj < 16; ++jj)
                sV[(dseg + jj) * VSTR + sr] = f2bf(vv[jj]);

            if (n0 + 32 < kvEnd) {      // prefetch next block -> global_prefetch_b8
                __builtin_prefetch(kp + 32 * Dc, 0, 1);
                __builtin_prefetch(vp + 32 * Dc, 0, 1);
            }
        }
        __syncthreads();   // staged data visible

        if (n0 <= wmax) {   // wave-uniform: EXEC stays all-ones for WMMA
            // ---- S = Q @ K^T : two 16x16 score tiles (cols 0..15 / 16..31) ----
            v8f s0 = vzero, s1 = vzero;
#pragma unroll
            for (int c = 0; c < 4; ++c) {
                const int doff = c * 32 + kh * 8;
                BFrag kb0, kb1;
                kb0.u[0] = *(const uint4*)&sK[(lm)      * KSTR + doff];
                kb0.u[1] = *(const uint4*)&sK[(lm)      * KSTR + doff + 16];
                kb1.u[0] = *(const uint4*)&sK[(16 + lm) * KSTR + doff];
                kb1.u[1] = *(const uint4*)&sK[(16 + lm) * KSTR + doff + 16];
                s0 = __builtin_amdgcn_wmma_f32_16x16x32_bf16(false, qf[c].v, false, kb0.v,
                                                             (short)0, s0, false, false);
                s1 = __builtin_amdgcn_wmma_f32_16x16x32_bf16(false, qf[c].v, false, kb1.v,
                                                             (short)0, s1, false, false);
            }

            // ---- online softmax (max only; sum comes from the ones-tile WMMA) ----
            unsigned short* Pw = &sP[w * PSZ];
#pragma unroll
            for (int r = 0; r < 8; ++r) {
                const int m = r + kh * 8;         // row within wave tile (C layout)
                float v0 = fmaf(s0[r], rscale, bias0[r]);
                float v1 = fmaf(s1[r], rscale, bias1[r]);
                if (n0 > thr0[r]) v0 = -1e30f;
                if (n0 > thr1[r]) v1 = -1e30f;
                bias0[r] += bstep;
                bias1[r] += bstep;
                float t = fmaxf(v0, v1);
                t = fmaxf(t, __shfl_xor(t, 1));
                t = fmaxf(t, __shfl_xor(t, 2));
                t = fmaxf(t, __shfl_xor(t, 4));
                t = fmaxf(t, __shfl_xor(t, 8));   // row max within 16-lane half
                const float nm   = fmaxf(rmax[r], t);
                const float corr = __expf(rmax[r] - nm);
                rmax[r] = nm;
                const float p0 = __expf(v0 - nm);
                const float p1 = __expf(v1 - nm);
#pragma unroll
                for (int jt = 0; jt < 9; ++jt) acc[jt][r] *= corr;
                // stage P (C layout -> LDS, row-major bf16)
                Pw[m * PSTR + lm]      = f2bf(p0);
                Pw[m * PSTR + 16 + lm] = f2bf(p1);
            }

            // ---- O += P @ V (8 d-tiles) and rowsum += P @ ones (tile 8) ----
            BFrag pf;
            pf.u[0] = *(const uint4*)&Pw[lm * PSTR + kh * 8];
            pf.u[1] = *(const uint4*)&Pw[lm * PSTR + 16 + kh * 8];
#pragma unroll
            for (int jt = 0; jt < 9; ++jt) {
                BFrag vb;
                vb.u[0] = *(const uint4*)&sV[(jt * 16 + lm) * VSTR + kh * 8];
                vb.u[1] = *(const uint4*)&sV[(jt * 16 + lm) * VSTR + 16 + kh * 8];
                acc[jt] = __builtin_amdgcn_wmma_f32_16x16x32_bf16(false, pf.v, false, vb.v,
                                                                  (short)0, acc[jt], false, false);
            }
        }
    }

    // ---- normalize and write O (fp32) ----
#pragma unroll
    for (int r = 0; r < 8; ++r) {
        // rowsum sits in column n=0 of the extra tile -> lane 0 (rows 0..7) / lane 16 (rows 8..15)
        const float rs  = __shfl(acc[8][r], kh << 4);
        const float inv = 1.0f / rs;
        const int m = r + kh * 8;
        const int i = qrow0 + m;
        float* op = O + base + (size_t)i * Dc + lm;
#pragma unroll
        for (int jt = 0; jt < 8; ++jt)
            op[jt * 16] = acc[jt][r] * inv;
    }
}

extern "C" void kernel_launch(void* const* d_in, const int* in_sizes, int n_in,
                              void* d_out, int out_size, void* d_ws, size_t ws_size,
                              hipStream_t stream) {
    const float* q = (const float*)d_in[0];
    const float* k = (const float*)d_in[1];
    const float* v = (const float*)d_in[2];
    // d_in[3] is the causal mask; causality is applied analytically in-kernel.
    float* out = (float*)d_out;
    (void)in_sizes; (void)n_in; (void)d_ws; (void)ws_size; (void)out_size;

    dim3 grid(Sc / 128, Bc * Hc);
    dim3 block(256);
    alibi_attn_kernel<<<grid, block, 0, stream>>>(q, k, v, out);
}
